// Net_76854144794847
// MI455X (gfx1250) — matrix operand order repaired and checked by reading, not compiled, not run on hardware
//
#include <hip/hip_runtime.h>
#include <stdint.h>

typedef __bf16 bf16;
typedef __attribute__((ext_vector_type(16))) __bf16 v16bf;
typedef __attribute__((ext_vector_type(8)))  __bf16 v8bf;
typedef __attribute__((ext_vector_type(4)))  __bf16 v4bf;
typedef __attribute__((ext_vector_type(8)))  float  v8f;
typedef __attribute__((ext_vector_type(4)))  int    v4i;

// async-LDS builtin takes typed int4 pointers in AS1 (global) / AS3 (LDS)
typedef __attribute__((address_space(1))) v4i v4i_as1;
typedef __attribute__((address_space(3))) v4i v4i_as3;

#if __has_builtin(__builtin_amdgcn_global_load_async_to_lds_b128) && \
    __has_builtin(__builtin_amdgcn_s_wait_asynccnt)
#define HAVE_ASYNC_LDS 1
#else
#define HAVE_ASYNC_LDS 0
#endif

static constexpr int   NN     = 8192;   // nodes
static constexpr int   F_IN   = 512;
static constexpr int   HID    = 128;
static constexpr int   NCLS   = 16;
static constexpr float ALPHA  = 0.8f;
static constexpr int   KSTEPS = 8;

// round-to-nearest-even f32 -> bf16 via bit ops (no reliance on __bf16 arith)
__device__ __forceinline__ bf16 f2bf(float f) {
  uint32_t u = __float_as_uint(f);
  uint32_t r = (u + 0x7FFFu + ((u >> 16) & 1u)) >> 16;
  uint16_t s = (uint16_t)r;
  bf16 b;
  __builtin_memcpy(&b, &s, sizeof(b));
  return b;
}

// ---------------------------------------------------------------------------
// dinv[i] = rsqrt(sum_j adj[i,j] + 1)   (self-loop included)
// ---------------------------------------------------------------------------
__global__ void k_degree(const float* __restrict__ adj, float* __restrict__ dinv) {
  __shared__ float red[256];
  const int row = blockIdx.x;
  const float* p = adj + (size_t)row * NN;
  float s = 0.f;
  for (int i = threadIdx.x; i < NN; i += 256) s += p[i];
  red[threadIdx.x] = s;
  __syncthreads();
  for (int off = 128; off > 0; off >>= 1) {
    if (threadIdx.x < off) red[threadIdx.x] += red[threadIdx.x + off];
    __syncthreads();
  }
  if (threadIdx.x == 0) dinv[row] = rsqrtf(fmaxf(red[0] + 1.0f, 1e-12f));
}

// ---------------------------------------------------------------------------
// adjn_bf16[i,j] = bf16( (adj[i,j] + (i==j)) * dinv[i] * dinv[j] )
// ---------------------------------------------------------------------------
__global__ void k_norm_cast(const float* __restrict__ adj,
                            const float* __restrict__ dinv,
                            bf16* __restrict__ adjn) {
  const size_t idx0 = ((size_t)blockIdx.x * 256 + threadIdx.x) * 8;
  const int i = (int)(idx0 >> 13);          // /8192
  const int j = (int)(idx0 & (NN - 1));     // %8192 (8-elem chunk stays in-row)
  const float di = dinv[i];
  const float4 a0 = *(const float4*)(adj + idx0);
  const float4 a1 = *(const float4*)(adj + idx0 + 4);
  const float v[8] = {a0.x, a0.y, a0.z, a0.w, a1.x, a1.y, a1.z, a1.w};
  v8bf o;
#pragma unroll
  for (int c = 0; c < 8; ++c) {
    float val = v[c] + ((j + c == i) ? 1.0f : 0.0f);
    o[c] = f2bf(val * di * dinv[j + c]);
  }
  *(v8bf*)(adjn + idx0) = o;
}

// ---------------------------------------------------------------------------
// f32 -> bf16 cast, row-major preserved (n multiple of 4)
// ---------------------------------------------------------------------------
__global__ void k_cast(const float* __restrict__ in, bf16* __restrict__ out, int n) {
  const int idx = (blockIdx.x * 256 + threadIdx.x) * 4;
  if (idx >= n) return;
  const float4 a = *(const float4*)(in + idx);
  v4bf o;
  o[0] = f2bf(a.x); o[1] = f2bf(a.y); o[2] = f2bf(a.z); o[3] = f2bf(a.w);
  *(v4bf*)(out + idx) = o;
}

// ---------------------------------------------------------------------------
// f32 (rows x cols) -> bf16 transposed (cols x rows), for small weights
// ---------------------------------------------------------------------------
__global__ void k_castT(const float* __restrict__ in, bf16* __restrict__ out,
                        int rows, int cols) {
  const int idx = blockIdx.x * 256 + threadIdx.x;
  if (idx >= rows * cols) return;
  const int r = idx / cols, c = idx % cols;
  out[(size_t)c * rows + r] = f2bf(in[idx]);
}

// ---------------------------------------------------------------------------
// bf16 WMMA GEMM with pre-transposed B:
//   acc = A(MxK, lda) @ BT(NxK, ldbt)^T
// epilogue:
//   v = s_acc*acc + (use_res ? s_res*Hres[m,n] : 0) + (use_bias ? bias[n] : 0)
//   v = do_relu ? max(v,0) : v
//   store_f32  -> Cf32[m*N+n]
//   store_bf16 -> bf16T ? Cbf16[n*ldcbf+m] (packed v8bf)  : Cbf16[m*N+n]
// Block = 256 threads = 8 wave32. BM=128 (16 rows/wave), BN in {64,16}.
// Double-buffered LDS staging; async global->LDS DMA when available.
// ---------------------------------------------------------------------------
template <int BN>
__global__ __launch_bounds__(256)
void k_gemm(const bf16* __restrict__ A, int lda,
            const bf16* __restrict__ BT, int ldbt,
            int K, int N,
            float s_acc, const float* __restrict__ Hres, float s_res, int use_res,
            const float* __restrict__ bias, int use_bias, int do_relu,
            float* __restrict__ Cf32, int store_f32,
            bf16* __restrict__ Cbf16, int store_bf16, int bf16T, int ldcbf) {
  constexpr int BM  = 128;
  constexpr int NT  = BN / 16;   // 16x16 accum tiles per wave
  constexpr int LDA = 40;        // padded LDS stride (bf16 elems): 80B rows

  __shared__ bf16 As[2 * BM * LDA];  // [buf][row][k]
  __shared__ bf16 Bs[2 * BN * LDA];  // [buf][n][k]

  const int tid  = threadIdx.x;
  const int wave = tid >> 5;
  const int lane = tid & 31;
  const int half = lane >> 4;    // 0: lanes 0-15, 1: lanes 16-31
  const int l16  = lane & 15;
  const int m0 = blockIdx.x * BM;
  const int n0 = blockIdx.y * BN;

  v8f acc[NT];
#pragma unroll
  for (int t = 0; t < NT; ++t)
#pragma unroll
    for (int r = 0; r < 8; ++r) acc[t][r] = 0.f;

  // A staging: thread t handles 16 bf16 of tile-row (t>>1), 16-col half (t&1)
  const int arow = tid >> 1;
  const int acol = (tid & 1) * 16;
  const bf16* agp = A + (size_t)(m0 + arow) * lda + acol;
  // B staging: thread t handles 8 bf16 of tile-row n (t>>2), k-quarter (t&3)
  const int brow = tid >> 2;
  const int bq   = (tid & 3) * 8;
  const bf16* bgp = BT + (size_t)(n0 + brow) * ldbt + bq;

  auto stage_tile = [&](int buf, int kk) {
    bf16* AsB = As + buf * (BM * LDA);
    bf16* BsB = Bs + buf * (BN * LDA);
    const bf16* asrc = agp + kk;
#if HAVE_ASYNC_LDS
    __builtin_amdgcn_global_load_async_to_lds_b128(
        (v4i_as1*)asrc, (v4i_as3*)(AsB + arow * LDA + acol), 0, 0);
    __builtin_amdgcn_global_load_async_to_lds_b128(
        (v4i_as1*)(asrc + 8), (v4i_as3*)(AsB + arow * LDA + acol + 8), 0, 0);
    if (BN == 64 || tid < 64) {
      __builtin_amdgcn_global_load_async_to_lds_b128(
          (v4i_as1*)(bgp + kk), (v4i_as3*)(BsB + brow * LDA + bq), 0, 0);
    }
#else
    const v8bf x0 = *(const v8bf*)(asrc);
    const v8bf x1 = *(const v8bf*)(asrc + 8);
    *(v8bf*)(AsB + arow * LDA + acol)     = x0;
    *(v8bf*)(AsB + arow * LDA + acol + 8) = x1;
    if (BN == 64 || tid < 64) {
      const v8bf y = *(const v8bf*)(bgp + kk);
      *(v8bf*)(BsB + brow * LDA + bq) = y;
    }
    __builtin_prefetch((const void*)(asrc + 32), 0, 1);  // global_prefetch_b8
#endif
  };

  stage_tile(0, 0);

  for (int k0 = 0; k0 < K; k0 += 32) {
    const int buf = (k0 >> 5) & 1;
#if HAVE_ASYNC_LDS
    __builtin_amdgcn_s_wait_asynccnt(0);   // this wave's DMA into `buf` landed
#endif
    __syncthreads();                        // publish `buf`; all reads of buf^1 done
    if (k0 + 32 < K) stage_tile(buf ^ 1, k0 + 32);  // overlaps with WMMAs below

    const bf16* AsB = As + buf * (BM * LDA);
    const bf16* BsB = Bs + buf * (BN * LDA);

    // A fragment (ISA 16-bit 16x32 layout): elem j -> K = (j>>3)*16 + half*8 + (j&7)
    const int ar = wave * 16 + l16;
    const v8bf alo = *(const v8bf*)(AsB + ar * LDA + half * 8);
    const v8bf ahi = *(const v8bf*)(AsB + ar * LDA + 16 + half * 8);
    v16bf a;
#pragma unroll
    for (int c = 0; c < 8; ++c) { a[c] = alo[c]; a[c + 8] = ahi[c]; }

    // load ALL B fragments first, then issue WMMAs back-to-back
    v16bf b[NT];
#pragma unroll
    for (int t = 0; t < NT; ++t) {
      const int bn = t * 16 + l16;   // elem j -> K = (j>>3)*16 + half*8 + (j&7), N = bn
      const v8bf blo = *(const v8bf*)(BsB + bn * LDA + half * 8);
      const v8bf bhi = *(const v8bf*)(BsB + bn * LDA + 16 + half * 8);
#pragma unroll
      for (int c = 0; c < 8; ++c) { b[t][c] = blo[c]; b[t][c + 8] = bhi[c]; }
    }
#pragma unroll
    for (int t = 0; t < NT; ++t)
      acc[t] = __builtin_amdgcn_wmma_f32_16x16x32_bf16(
          false, a, false, b[t], (short)0, acc[t], false, false);
  }

  // epilogue (C layout: VGPR r, lanes0-15 -> M=r, lanes16-31 -> M=8+r, N=l16)
  const int mb = m0 + wave * 16 + half * 8;
#pragma unroll
  for (int t = 0; t < NT; ++t) {
    const int n = n0 + t * 16 + l16;
    float vv[8];
#pragma unroll
    for (int r = 0; r < 8; ++r) {
      const size_t off = (size_t)(mb + r) * N + n;
      float v = s_acc * acc[t][r];
      if (use_res)  v += s_res * Hres[off];
      if (use_bias) v += bias[n];
      if (do_relu)  v = fmaxf(v, 0.f);
      if (store_f32) Cf32[off] = v;
      vv[r] = v;
    }
    if (store_bf16) {
      if (bf16T) {  // transposed: one packed 16B store per lane per tile
        v8bf o;
#pragma unroll
        for (int r = 0; r < 8; ++r) o[r] = f2bf(vv[r]);
        *(v8bf*)(Cbf16 + (size_t)n * ldcbf + mb) = o;
      } else {      // row-major (used once, for the relu activation)
#pragma unroll
        for (int r = 0; r < 8; ++r)
          Cbf16[(size_t)(mb + r) * N + n] = f2bf(vv[r]);
      }
    }
  }
}

// ---------------------------------------------------------------------------
// in-place row-wise log_softmax over 16 classes, one thread per row
// ---------------------------------------------------------------------------
__global__ void k_logsoftmax(float* __restrict__ out) {
  const int row = blockIdx.x * 256 + threadIdx.x;
  float* p = out + (size_t)row * NCLS;
  float v[NCLS];
  float m = -3.4e38f;
#pragma unroll
  for (int i = 0; i < NCLS; ++i) { v[i] = p[i]; m = fmaxf(m, v[i]); }
  float s = 0.f;
#pragma unroll
  for (int i = 0; i < NCLS; ++i) s += expf(v[i] - m);
  const float lse = logf(s) + m;
#pragma unroll
  for (int i = 0; i < NCLS; ++i) p[i] = v[i] - lse;
}

// ---------------------------------------------------------------------------
extern "C" void kernel_launch(void* const* d_in, const int* in_sizes, int n_in,
                              void* d_out, int out_size, void* d_ws, size_t ws_size,
                              hipStream_t stream) {
  (void)in_sizes; (void)n_in; (void)out_size; (void)ws_size;
  const float* x   = (const float*)d_in[0];
  const float* adj = (const float*)d_in[1];
  const float* W1  = (const float*)d_in[2];
  const float* b1  = (const float*)d_in[3];
  const float* W2  = (const float*)d_in[4];
  const float* b2  = (const float*)d_in[5];
  float* out = (float*)d_out;

  char* ws = (char*)d_ws;
  size_t off = 0;
  auto alloc = [&](size_t bytes) -> void* {
    void* p = ws + off;
    off = (off + bytes + 255) & ~(size_t)255;
    return p;
  };
  bf16*  adjn    = (bf16*) alloc((size_t)NN * NN * 2);   // 134 MB, L2-resident
  bf16*  xbf     = (bf16*) alloc((size_t)NN * F_IN * 2);
  bf16*  w1T     = (bf16*) alloc((size_t)HID * F_IN * 2);   // 128 x 512
  bf16*  w2T     = (bf16*) alloc((size_t)NCLS * HID * 2);   // 16 x 128
  float* dinv    = (float*)alloc((size_t)NN * 4);
  float* h1      = (float*)alloc((size_t)NN * HID * 4);
  bf16*  pTA     = (bf16*) alloc((size_t)HID * NN * 2);     // out_k^T ping
  bf16*  pTB     = (bf16*) alloc((size_t)HID * NN * 2);     // out_k^T pong
  bf16*  relu_bf = (bf16*) alloc((size_t)NN * HID * 2);     // row-major relu(h)
  float* h2      = (float*)alloc((size_t)NN * NCLS * 4);
  bf16*  qTA     = (bf16*) alloc((size_t)NCLS * NN * 2);
  bf16*  qTB     = (bf16*) alloc((size_t)NCLS * NN * 2);

  const float c_res  = 1.0f / (1.0f + ALPHA);
  const float c_prop = ALPHA / (1.0f + ALPHA);

  // 1) normalization
  k_degree<<<NN, 256, 0, stream>>>(adj, dinv);
  k_norm_cast<<<(unsigned)(((size_t)NN * NN) / (256 * 8)), 256, 0, stream>>>(adj, dinv, adjn);

  // 2) bf16 casts (weights transposed to N x K for the GEMM's B operand)
  k_cast<<<(NN * F_IN) / 1024, 256, 0, stream>>>(x, xbf, NN * F_IN);
  k_castT<<<(F_IN * HID + 255) / 256, 256, 0, stream>>>(W1, w1T, F_IN, HID);
  k_castT<<<(HID * NCLS + 255) / 256, 256, 0, stream>>>(W2, w2T, HID, NCLS);

  // 3) layer 1: h1 = x @ W1  (f32 residual, bf16 transposed out_0)
  k_gemm<64><<<dim3(NN / 128, HID / 64), 256, 0, stream>>>(
      xbf, F_IN, w1T, F_IN, F_IN, HID,
      1.0f, h1, 0.0f, 0, b1, 0, 0,
      h1, 1, pTA, 1, /*bf16T=*/1, NN);

  // 4) layer-1 diffusion: 8 x (c_prop * A_n@out + c_res * h1); last: +b1, relu
  bf16* cur = pTA; bf16* nxt = pTB;
  for (int s = 0; s < KSTEPS; ++s) {
    const int last = (s == KSTEPS - 1);
    k_gemm<64><<<dim3(NN / 128, HID / 64), 256, 0, stream>>>(
        adjn, NN, cur, NN, NN, HID,
        c_prop, h1, c_res, 1,
        b1, last, last,
        nullptr, 0,
        last ? relu_bf : nxt, 1, /*bf16T=*/last ? 0 : 1, NN);
    bf16* t = cur; cur = nxt; nxt = t;
  }

  // 5) layer 2: h2 = relu(h) @ W2  (f32 residual, bf16 transposed out_0)
  k_gemm<16><<<dim3(NN / 128, 1), 256, 0, stream>>>(
      relu_bf, HID, w2T, HID, HID, NCLS,
      1.0f, h2, 0.0f, 0, b2, 0, 0,
      h2, 1, qTA, 1, /*bf16T=*/1, NN);

  // 6) layer-2 diffusion; last step: +b2, f32 logits straight into d_out
  bf16* c2 = qTA; bf16* n2 = qTB;
  for (int s = 0; s < KSTEPS; ++s) {
    const int last = (s == KSTEPS - 1);
    k_gemm<16><<<dim3(NN / 128, 1), 256, 0, stream>>>(
        adjn, NN, c2, NN, NN, NCLS,
        c_prop, h2, c_res, 1,
        b2, last, 0,
        last ? out : nullptr, last,
        n2, last ? 0 : 1, /*bf16T=*/1, NN);
    bf16* t = c2; c2 = n2; n2 = t;
  }

  // 7) log_softmax in place on d_out
  k_logsoftmax<<<NN / 256, 256, 0, stream>>>(out);
}
